// LinearAttention_28681791602716
// MI455X (gfx1250) — matrix operand order repaired and checked by reading, not compiled, not run on hardware
//
#include <hip/hip_runtime.h>
#include <hip/hip_bf16.h>
#include <hip/hip_fp16.h>

typedef __attribute__((ext_vector_type(16))) _Float16 v16h;
typedef __attribute__((ext_vector_type(8)))  _Float16 v8h;
typedef __attribute__((ext_vector_type(8)))  float    v8f;

#define SEQ     2048
#define DMODEL  1024
#define NHEADS  16
#define DHEAD   64
#define CHUNK   32
#define NPART   16                 // partitions per head for the parallel scan
#define CPP     (SEQ / CHUNK / NPART)  // chunks per partition = 4

// GEMM block tile
#define GM 64
#define GN 128
#define GK 32

// ---------- fragment helpers (wave32 WMMA operands; contiguous k, same order in A and B) ----------

// A-matrix 16x32 f16: lane L -> M = L&15, half = L>>4.
// elems j=0..15 -> K = 16*half + j (contiguous)
__device__ __forceinline__ v16h load_a16(const _Float16* base, int ld, int row0,
                                         int k0, unsigned lane) {
  const unsigned m = lane & 15u, h = lane >> 4;
  const _Float16* p = base + (size_t)(row0 + (int)m) * (size_t)ld + k0;
  v8h lo = *(const v8h*)(p + 16u * h);
  v8h hi = *(const v8h*)(p + 16u * h + 8u);
  v16h r;
#pragma unroll
  for (int j = 0; j < 8; ++j) { r[j] = lo[j]; r[j + 8] = hi[j]; }
  return r;
}

// B-matrix 32x16 f16 with B[k][n] = base[n][k] (row-per-N storage, i.e. W[N][K]):
// lane L -> N = L&15, half = L>>4; elems j=0..15 -> K = 16*half + j (contiguous).
__device__ __forceinline__ v16h load_b16(const _Float16* base, int ld, int n0,
                                         int k0, unsigned lane) {
  const unsigned n = lane & 15u, h = lane >> 4;
  const _Float16* p = base + (size_t)(n0 + (int)n) * (size_t)ld + k0 + 16u * h;
  v8h lo = *(const v8h*)(p);
  v8h hi = *(const v8h*)(p + 8);
  v16h r;
#pragma unroll
  for (int j = 0; j < 8; ++j) { r[j] = lo[j]; r[j + 8] = hi[j]; }
  return r;
}

__device__ __forceinline__ v8f wmma_f16(v16h a, v16h b, v8f c) {
  return __builtin_amdgcn_wmma_f32_16x16x32_f16(false, a, false, b,
                                                (short)0, c, false, false);
}

__device__ __forceinline__ v8f zero8() {
  v8f z;
#pragma unroll
  for (int i = 0; i < 8; ++i) z[i] = 0.f;
  return z;
}

__device__ __forceinline__ float softplus_f(float x) {
  return (x > 20.f) ? x : log1pf(__expf(x));
}

// async copy of 16 bytes global -> LDS (per lane), tracked by ASYNCcnt
__device__ __forceinline__ void async_cp16(const _Float16* gsrc, _Float16* ldst) {
  const unsigned lo = (unsigned)(uintptr_t)ldst;  // flat->LDS uses addr[31:0]
  asm volatile("global_load_async_to_lds_b128 %0, %1, off"
               :: "v"(lo), "v"((unsigned long long)(uintptr_t)gsrc)
               : "memory");
}

// ---------- f32 -> f16 convert ----------
__global__ void f32_to_f16_kernel(const float* __restrict__ src,
                                  _Float16* __restrict__ dst, int n) {
  int i = blockIdx.x * blockDim.x + threadIdx.x;
  int stride = gridDim.x * blockDim.x;
  for (; i < n; i += stride) dst[i] = (_Float16)src[i];
}

// ---------- GEMM: C = act(A @ W^T + bias) ----------
// A: [M x K] f16 row-major; W: [N x K] f16 row-major (torch Linear weight).
// Block 256 thr = 8 waves (2M x 4N); each wave owns a 32x32 patch (2x2 WMMA tiles).
// A/B tiles staged to LDS with double-buffered async global->LDS copies.
template <bool SOFTPLUS, bool F16OUT>
__global__ __launch_bounds__(256) void gemm_kernel(
    const _Float16* __restrict__ A, const _Float16* __restrict__ W,
    const float* __restrict__ bias, _Float16* __restrict__ outH,
    float* __restrict__ outF, int N, int K) {
  __shared__ __align__(16) _Float16 Abuf[2][GM * GK];
  __shared__ __align__(16) _Float16 Bbuf[2][GN * GK];
  const unsigned t = threadIdx.x;
  const unsigned lane = t & 31u, wave = t >> 5;
  const int mw = (int)(wave >> 2), nw = (int)(wave & 3u);
  const int blockM = blockIdx.x * GM;
  const int blockN = blockIdx.y * GN;

  v8f acc[2][2];
#pragma unroll
  for (int m = 0; m < 2; ++m)
#pragma unroll
    for (int n = 0; n < 2; ++n) acc[m][n] = zero8();

  // Stage one K-slab: A tile 64x32 (256 x 16B chunks) + B tile 128x32 (512 chunks).
  // Each thread issues 3 async 16B copies -> 3 ASYNC instructions per wave per slab.
  auto stage = [&](int buf, int k0) {
#pragma unroll
    for (int cc = 0; cc < 3; ++cc) {
      const int chunk = (int)t + cc * 256;
      if (chunk < 256) {
        const int row = chunk >> 2, seg = chunk & 3;
        async_cp16(A + (size_t)(blockM + row) * K + k0 + seg * 8,
                   &Abuf[buf][row * GK + seg * 8]);
      } else {
        const int c2 = chunk - 256;
        const int row = c2 >> 2, seg = c2 & 3;
        async_cp16(W + (size_t)(blockN + row) * K + k0 + seg * 8,
                   &Bbuf[buf][row * GK + seg * 8]);
      }
    }
  };

  const int KT = K / GK;
  stage(0, 0);
  for (int kt = 0; kt < KT; ++kt) {
    const int buf = kt & 1;
    if (kt + 1 < KT) {
      stage(buf ^ 1, (kt + 1) * GK);
      asm volatile("s_wait_asynccnt 0x3" ::: "memory");  // current slab done
    } else {
      asm volatile("s_wait_asynccnt 0x0" ::: "memory");
    }
    __syncthreads();
    v16h a0 = load_a16(Abuf[buf], GK, mw * 32 + 0, 0, lane);
    v16h a1 = load_a16(Abuf[buf], GK, mw * 32 + 16, 0, lane);
    v16h b0 = load_b16(Bbuf[buf], GK, nw * 32 + 0, 0, lane);
    v16h b1 = load_b16(Bbuf[buf], GK, nw * 32 + 16, 0, lane);
    acc[0][0] = wmma_f16(a0, b0, acc[0][0]);
    acc[0][1] = wmma_f16(a0, b1, acc[0][1]);
    acc[1][0] = wmma_f16(a1, b0, acc[1][0]);
    acc[1][1] = wmma_f16(a1, b1, acc[1][1]);
    __syncthreads();  // reads done before this buffer is restaged
  }

  const unsigned ln = lane & 15u, lh = lane >> 4;
#pragma unroll
  for (int m = 0; m < 2; ++m) {
#pragma unroll
    for (int n = 0; n < 2; ++n) {
      const int col = blockN + nw * 32 + n * 16 + (int)ln;
      const float bval = bias[col];
#pragma unroll
      for (int r = 0; r < 8; ++r) {
        const int row = blockM + mw * 32 + m * 16 + (int)(8u * lh) + r;
        float v = acc[m][n][r] + bval;
        if (SOFTPLUS) v = softplus_f(v);
        if (F16OUT) outH[(size_t)row * N + col] = (_Float16)v;
        else        outF[(size_t)row * N + col] = v;
      }
    }
  }
}

// ---------- pass A: per-partition state sums  KVp = K^T V, k1p = colsum(K) ----------
__global__ __launch_bounds__(128) void attn_state_kernel(
    const _Float16* __restrict__ Kh, const _Float16* __restrict__ Vh,
    float* __restrict__ KVp, float* __restrict__ k1p) {
  const int p = blockIdx.x, head = blockIdx.y;
  const unsigned t = threadIdx.x, lane = t & 31u, wave = t >> 5;
  const int hoff = head * DHEAD;
  __shared__ __align__(16) _Float16 Kt[DHEAD * CHUNK];  // [D][s]
  __shared__ __align__(16) _Float16 Vt[DHEAD * CHUNK];  // [e][s]

  v8f kv[4];
#pragma unroll
  for (int q = 0; q < 4; ++q) kv[q] = zero8();
  float k1acc = 0.f;
  const unsigned ln = lane & 15u, lh = lane >> 4;

  for (int c = 0; c < CPP; ++c) {
    const int s0 = (p * CPP + c) * CHUNK;
    for (int idx = (int)t; idx < DHEAD * CHUNK; idx += 128) {
      int e = idx & 63, i = idx >> 6;
      Kt[e * CHUNK + i] = Kh[(size_t)(s0 + i) * DMODEL + hoff + e];
      Vt[e * CHUNK + i] = Vh[(size_t)(s0 + i) * DMODEL + hoff + e];
    }
    __syncthreads();
    v16h ak = load_a16(Kt, CHUNK, (int)wave * 16, 0, lane);  // A[D][s]
#pragma unroll
    for (int ne = 0; ne < 4; ++ne) {
      v16h bv = load_b16(Vt, CHUNK, ne * 16, 0, lane);       // B[s][e]
      kv[ne] = wmma_f16(ak, bv, kv[ne]);
    }
    if (t < DHEAD) {
      for (int i2 = 0; i2 < CHUNK; ++i2) k1acc += (float)Kt[t * CHUNK + i2];
    }
    __syncthreads();
  }

  float* dst = KVp + (size_t)(head * NPART + p) * (DHEAD * DHEAD);  // [D][e]
#pragma unroll
  for (int ne = 0; ne < 4; ++ne) {
#pragma unroll
    for (int r = 0; r < 8; ++r) {
      int D = 16 * (int)wave + (int)(8u * lh) + r;
      int e = 16 * ne + (int)ln;
      dst[D * DHEAD + e] = kv[ne][r];
    }
  }
  if (t < DHEAD) k1p[(size_t)(head * NPART + p) * DHEAD + t] = k1acc;
}

// ---------- pass B: exclusive prefix over partitions ----------
__global__ __launch_bounds__(256) void attn_scan_kernel(
    const float* __restrict__ KVp, const float* __restrict__ k1p,
    float* __restrict__ KVs, float* __restrict__ k1s) {
  const int head = blockIdx.x;
  const int t = (int)threadIdx.x;
  for (int e = t; e < DHEAD * DHEAD; e += 256) {
    float run = 0.f;
    for (int p = 0; p < NPART; ++p) {
      size_t idx = (size_t)(head * NPART + p) * (DHEAD * DHEAD) + e;
      KVs[idx] = run;
      run += KVp[idx];
    }
  }
  if (t < DHEAD) {
    float run = 0.f;
    for (int p = 0; p < NPART; ++p) {
      size_t idx = (size_t)(head * NPART + p) * DHEAD + t;
      k1s[idx] = run;
      run += k1p[idx];
    }
  }
}

// ---------- pass C: chunked causal linear attention within a partition ----------
__global__ __launch_bounds__(128) void attn_out_kernel(
    const _Float16* __restrict__ Qh, const _Float16* __restrict__ Kh,
    const _Float16* __restrict__ Vh, const float* __restrict__ KVs,
    const float* __restrict__ k1s, _Float16* __restrict__ Oh) {
  const int p = blockIdx.x, head = blockIdx.y;
  const unsigned t = threadIdx.x, lane = t & 31u, wave = t >> 5;
  const int hoff = head * DHEAD;

  __shared__ __align__(16) _Float16 KVhT[DHEAD * DHEAD];  // [e][D] f16 mirror of KV
  __shared__ __align__(16) _Float16 Kt[DHEAD * CHUNK];    // [D][s]
  __shared__ __align__(16) _Float16 Vt[DHEAD * CHUNK];    // [e][s]
  __shared__ __align__(16) _Float16 Sm[CHUNK * CHUNK];    // masked S, [i][j]
  __shared__ float k1[DHEAD];
  __shared__ float den[CHUNK];

  const unsigned ln = lane & 15u, lh = lane >> 4;

  // init state from partition prefix
  const float* kvin = KVs + (size_t)(head * NPART + p) * (DHEAD * DHEAD);
  v8f kv[4];  // wave owns KV rows D in [16*wave, 16*wave+16)
#pragma unroll
  for (int ne = 0; ne < 4; ++ne) {
#pragma unroll
    for (int r = 0; r < 8; ++r) {
      int D = 16 * (int)wave + (int)(8u * lh) + r;
      int e = 16 * ne + (int)ln;
      float v0 = kvin[D * DHEAD + e];
      kv[ne][r] = v0;
      KVhT[e * DHEAD + D] = (_Float16)v0;
    }
  }
  if (t < DHEAD) k1[t] = k1s[(size_t)(head * NPART + p) * DHEAD + t];
  __syncthreads();

  for (int c = 0; c < CPP; ++c) {
    const int s0 = (p * CPP + c) * CHUNK;

    // ---- phase 1: per-wave S tile + stage transposed K/V chunk ----
    {
      const int mt = (int)(wave >> 1), nt = (int)(wave & 1u);
      v8f s = zero8();
      v16h a0 = load_a16(Qh + hoff, DMODEL, s0 + mt * 16, 0, lane);
      v16h b0 = load_b16(Kh + hoff, DMODEL, s0 + nt * 16, 0, lane);
      s = wmma_f16(a0, b0, s);
      v16h a1 = load_a16(Qh + hoff, DMODEL, s0 + mt * 16, 32, lane);
      v16h b1 = load_b16(Kh + hoff, DMODEL, s0 + nt * 16, 32, lane);
      s = wmma_f16(a1, b1, s);
#pragma unroll
      for (int r = 0; r < 8; ++r) {
        int mi = 16 * mt + (int)(8u * lh) + r;
        int nj = 16 * nt + (int)ln;
        Sm[mi * CHUNK + nj] = (_Float16)((mi >= nj) ? s[r] : 0.f);
      }
    }
    for (int idx = (int)t; idx < DHEAD * CHUNK; idx += 128) {
      int e = idx & 63, i = idx >> 6;
      Kt[e * CHUNK + i] = Kh[(size_t)(s0 + i) * DMODEL + hoff + e];
      Vt[e * CHUNK + i] = Vh[(size_t)(s0 + i) * DMODEL + hoff + e];
    }
    __syncthreads();

    // ---- phase 2: denominator + O tiles (read KVhT/k1 state) ----
    if (t < CHUNK) {
      float d = 0.f;
      for (int j = 0; j < CHUNK; ++j) d += (float)Sm[t * CHUNK + j];
      for (int D = 0; D < DHEAD; ++D)
        d += (float)Qh[(size_t)(s0 + (int)t) * DMODEL + hoff + D] * k1[D];
      den[t] = d;
    }
    v8f o[2];
    {
      v16h bk0 = load_b16(KVhT, DHEAD, (int)wave * 16, 0, lane);
      v16h bk1 = load_b16(KVhT, DHEAD, (int)wave * 16, 32, lane);
      v16h bvv = load_b16(Vt, CHUNK, (int)wave * 16, 0, lane);
#pragma unroll
      for (int m = 0; m < 2; ++m) {
        v8f a = zero8();
        v16h q0 = load_a16(Qh + hoff, DMODEL, s0 + m * 16, 0, lane);
        a = wmma_f16(q0, bk0, a);  // inter: Q * KV_prev
        v16h q1 = load_a16(Qh + hoff, DMODEL, s0 + m * 16, 32, lane);
        a = wmma_f16(q1, bk1, a);
        v16h as = load_a16(Sm, CHUNK, m * 16, 0, lane);
        a = wmma_f16(as, bvv, a);  // intra: S_mask * V
        o[m] = a;
      }
    }
    __syncthreads();  // den ready; all reads of old state done

#pragma unroll
    for (int m = 0; m < 2; ++m) {
#pragma unroll
      for (int r = 0; r < 8; ++r) {
        int li = 16 * m + (int)(8u * lh) + r;
        float v = o[m][r] / den[li];
        Oh[(size_t)(s0 + li) * DMODEL + hoff + (int)wave * 16 + (int)ln] =
            (_Float16)v;
      }
    }

    // ---- phase 3: state update KV += K^T V ; k1 += colsum(K) ----
    {
      v16h ak = load_a16(Kt, CHUNK, (int)wave * 16, 0, lane);
#pragma unroll
      for (int ne = 0; ne < 4; ++ne) {
        v16h bv = load_b16(Vt, CHUNK, ne * 16, 0, lane);
        kv[ne] = wmma_f16(ak, bv, kv[ne]);
#pragma unroll
        for (int r = 0; r < 8; ++r) {
          KVhT[(16 * ne + (int)ln) * DHEAD + 16 * (int)wave + (int)(8u * lh) + r] =
              (_Float16)kv[ne][r];
        }
      }
    }
    if (t < DHEAD) {
      float ssum = 0.f;
      for (int i2 = 0; i2 < CHUNK; ++i2) ssum += (float)Kt[t * CHUNK + i2];
      k1[t] += ssum;
    }
    __syncthreads();
  }
}

// ---------- host-side launch ----------
extern "C" void kernel_launch(void* const* d_in, const int* in_sizes, int n_in,
                              void* d_out, int out_size, void* d_ws,
                              size_t ws_size, hipStream_t stream) {
  (void)in_sizes; (void)n_in; (void)out_size; (void)ws_size;
  const float* x  = (const float*)d_in[0];
  const float* Wq = (const float*)d_in[1];
  const float* bq = (const float*)d_in[2];
  const float* Wk = (const float*)d_in[3];
  const float* bk = (const float*)d_in[4];
  const float* Wv = (const float*)d_in[5];
  const float* bv = (const float*)d_in[6];
  const float* Wo = (const float*)d_in[7];
  const float* bo = (const float*)d_in[8];
  float* out = (float*)d_out;

  char* ws = (char*)d_ws;
  size_t off = 0;
  auto carve = [&](size_t bytes) -> void* {
    void* p = (void*)(ws + off);
    off += (bytes + 255) & ~(size_t)255;
    return p;
  };
  const size_t XB = (size_t)SEQ * DMODEL * sizeof(_Float16);
  const size_t WB = (size_t)DMODEL * DMODEL * sizeof(_Float16);
  const size_t SB = (size_t)NHEADS * NPART * DHEAD * DHEAD * sizeof(float);
  const size_t KB = (size_t)NHEADS * NPART * DHEAD * sizeof(float);
  _Float16* xh  = (_Float16*)carve(XB);
  _Float16* Wqh = (_Float16*)carve(WB);
  _Float16* Wkh = (_Float16*)carve(WB);
  _Float16* Wvh = (_Float16*)carve(WB);
  _Float16* Woh = (_Float16*)carve(WB);
  _Float16* Qf  = (_Float16*)carve(XB);  // softplus(q), f16
  _Float16* Kf  = (_Float16*)carve(XB);  // softplus(k), f16
  _Float16* Vf  = (_Float16*)carve(XB);
  _Float16* Af  = (_Float16*)carve(XB);  // attention output, f16
  float* KVp = (float*)carve(SB);
  float* KVs = (float*)carve(SB);
  float* k1p = (float*)carve(KB);
  float* k1s = (float*)carve(KB);

  const int cthr = 256;
  f32_to_f16_kernel<<<(SEQ * DMODEL + cthr - 1) / cthr, cthr, 0, stream>>>(
      x, xh, SEQ * DMODEL);
  f32_to_f16_kernel<<<(DMODEL * DMODEL + cthr - 1) / cthr, cthr, 0, stream>>>(
      Wq, Wqh, DMODEL * DMODEL);
  f32_to_f16_kernel<<<(DMODEL * DMODEL + cthr - 1) / cthr, cthr, 0, stream>>>(
      Wk, Wkh, DMODEL * DMODEL);
  f32_to_f16_kernel<<<(DMODEL * DMODEL + cthr - 1) / cthr, cthr, 0, stream>>>(
      Wv, Wvh, DMODEL * DMODEL);
  f32_to_f16_kernel<<<(DMODEL * DMODEL + cthr - 1) / cthr, cthr, 0, stream>>>(
      Wo, Woh, DMODEL * DMODEL);

  dim3 gg(SEQ / GM, DMODEL / GN);
  gemm_kernel<true, true><<<gg, 256, 0, stream>>>(xh, Wqh, bq, Qf, nullptr,
                                                  DMODEL, DMODEL);
  gemm_kernel<true, true><<<gg, 256, 0, stream>>>(xh, Wkh, bk, Kf, nullptr,
                                                  DMODEL, DMODEL);
  gemm_kernel<false, true><<<gg, 256, 0, stream>>>(xh, Wvh, bv, Vf, nullptr,
                                                   DMODEL, DMODEL);

  dim3 ga(NPART, NHEADS);
  attn_state_kernel<<<ga, 128, 0, stream>>>(Kf, Vf, KVp, k1p);
  attn_scan_kernel<<<NHEADS, 256, 0, stream>>>(KVp, k1p, KVs, k1s);
  attn_out_kernel<<<ga, 128, 0, stream>>>(Qf, Kf, Vf, KVs, k1s, Af);

  gemm_kernel<false, false><<<gg, 256, 0, stream>>>(Af, Woh, bo, nullptr, out,
                                                    DMODEL, DMODEL);
}
